// DimeNetPlusPlusWrap_11321533792784
// MI455X (gfx1250) — matrix operand order repaired and checked by reading, not compiled, not run on hardware
//
#include <hip/hip_runtime.h>
#include <hip/hip_bf16.h>

// ---------------------------------------------------------------------------
// DimeNet++ interaction block for MI455X (gfx1250, wave32, WMMA bf16 path)
// ---------------------------------------------------------------------------

#define E_EDGES 200000
#define T_TRIP  2000000
#define H_DIM   128
#define I_DIM   64
#define B_DIM   8
#define NR_DIM  6
#define NSR_DIM 42   // NS*NR = 7*6

typedef __attribute__((ext_vector_type(16))) __bf16 v16bf;
typedef __attribute__((ext_vector_type(8)))  float  v8f;

__device__ __forceinline__ float silu_f(float v) {
    return v / (1.0f + __expf(-v));
}

__device__ __forceinline__ void zero_acc(v8f* a, int n) {
    for (int i = 0; i < n; ++i)
        for (int j = 0; j < 8; ++j) a[i][j] = 0.0f;
}

// 16-row x NT*16-col tile GEMM via v_wmma_f32_16x16x32_bf16.
// A: [16 x K] bf16, row-major, lda elements per row (LDS resident).
// B: [K x NT*16] bf16, row-major, ldb elements per row (global, L2 resident).
// Fragment layouts per CDNA5 ISA 7.12.2:
//   A (16-bit 16x32): lane half h, row r=lane&15; VGPR j<4 -> K=k0+8h+2j(+1),
//                     VGPR j>=4 -> K=k0+8h+16+2(j-4)(+1)
//   B: lane half h, col N=lane&15; elem s<8 -> K=k0+8h+s, s>=8 -> K=k0+16+8h+(s-8)
//   C/D: VGPR v at lane l -> M = v + 8*(l>>4), N = l&15
template<int K, int NT>
__device__ __forceinline__ void mm_tile(const __bf16* A, int lda,
                                        const __bf16* B, int ldb,
                                        v8f* acc) {
    const int lane = threadIdx.x & 31;
    const int r16  = lane & 15;
    const int hh   = lane >> 4;
#pragma unroll
    for (int k0 = 0; k0 < K; k0 += 32) {
        union { v16bf v; unsigned u[8]; } a;
        const unsigned* arow = (const unsigned*)(A + r16 * lda + k0 + 8 * hh);
#pragma unroll
        for (int j = 0; j < 4; ++j) {
            a.u[j]     = arow[j];       // K = k0+8h + {0..7}
            a.u[4 + j] = arow[8 + j];   // K = k0+8h+16 + {0..7}
        }
#pragma unroll
        for (int nt = 0; nt < NT; ++nt) {
            union { v16bf v; __bf16 e[16]; } b;
            const __bf16* bcol = B + (size_t)(k0 + 8 * hh) * ldb + nt * 16 + r16;
#pragma unroll
            for (int s = 0; s < 8; ++s) {
                b.e[s]     = bcol[(size_t)s * ldb];
                b.e[8 + s] = bcol[(size_t)(16 + s) * ldb];
            }
            acc[nt] = __builtin_amdgcn_wmma_f32_16x16x32_bf16(
                false, a.v, false, b.v, (short)0, acc[nt], false, false);
        }
    }
}

// ---------------------------------------------------------------------------
// Prep kernels
// ---------------------------------------------------------------------------

__global__ void cvt_bf16_kernel(const float* __restrict__ src,
                                __bf16* __restrict__ dst, int n) {
    int i = blockIdx.x * blockDim.x + threadIdx.x;
    if (i < n) dst[i] = (__bf16)src[i];
}

// W_rbf_c[6][128] = W_rbf1[6][8] @ W_rbf2[8][128]
__global__ void combine_rbf_kernel(const float* __restrict__ W1,
                                   const float* __restrict__ W2,
                                   float* __restrict__ out) {
    int i = blockIdx.x * blockDim.x + threadIdx.x;
    if (i < NR_DIM * H_DIM) {
        int r = i >> 7, c = i & 127;
        float s = 0.0f;
#pragma unroll
        for (int b = 0; b < B_DIM; ++b) s += W1[r * B_DIM + b] * W2[b * H_DIM + c];
        out[i] = s;
    }
}

// ---------------------------------------------------------------------------
// edge_front: x_ji = silu(x@W_ji+b_ji)
//             t    = silu(x@W_kj+b_kj) * (rbf @ W_rbf_c)
//             x_kj_down = silu(t @ W_down)
// 128 threads (4 waves), 64 edges per block, 16 edges per wave.
// ---------------------------------------------------------------------------
__global__ void edge_front_kernel(const float* __restrict__ x,
                                  const float* __restrict__ rbf,
                                  const __bf16* __restrict__ Wji,
                                  const __bf16* __restrict__ Wkj,
                                  const __bf16* __restrict__ Wdn,
                                  const float* __restrict__ b_ji,
                                  const float* __restrict__ b_kj,
                                  const float* __restrict__ Wrbfc,
                                  float* __restrict__ x_ji_out,
                                  float* __restrict__ xkj_down_out,
                                  int E) {
    __shared__ __bf16 sX[64 * 128];
    __shared__ float  sRbf[64 * NR_DIM];
    __shared__ float  sWr[NR_DIM * 128];

    const int tid  = threadIdx.x;
    const int e0   = blockIdx.x * 64;
    const int wave = tid >> 5, lane = tid & 31;
    const int r16  = lane & 15, hh = lane >> 4;

    // stage x tile (f32 -> bf16), rbf tile, combined rbf weight
    for (int i = tid; i < 64 * 128; i += 128) {
        int r = i >> 7, c = i & 127;
        int ge = e0 + r;
        float v = (ge < E) ? x[(size_t)ge * H_DIM + c] : 0.0f;
        sX[i] = (__bf16)v;
    }
    for (int i = tid; i < 64 * NR_DIM; i += 128) {
        int r = i / NR_DIM, c = i % NR_DIM;
        int ge = e0 + r;
        sRbf[i] = (ge < E) ? rbf[(size_t)ge * NR_DIM + c] : 0.0f;
    }
    for (int i = tid; i < NR_DIM * 128; i += 128) sWr[i] = Wrbfc[i];
    __syncthreads();

    const __bf16* Aslice = sX + wave * 16 * 128;

    // ---- GEMM 1: x @ W_ji -> x_ji ----
    v8f acc[8];
    zero_acc(acc, 8);
    mm_tile<128, 8>(Aslice, 128, Wji, 128, acc);
#pragma unroll
    for (int nt = 0; nt < 8; ++nt) {
        int n = nt * 16 + r16;
        float bb = b_ji[n];
#pragma unroll
        for (int v = 0; v < 8; ++v) {
            int ge = e0 + wave * 16 + v + 8 * hh;
            if (ge < E) x_ji_out[(size_t)ge * H_DIM + n] = silu_f(acc[nt][v] + bb);
        }
    }

    // ---- GEMM 2: x @ W_kj, scale by rbf embedding, write t back into sX ----
    zero_acc(acc, 8);
    mm_tile<128, 8>(Aslice, 128, Wkj, 128, acc);
#pragma unroll
    for (int nt = 0; nt < 8; ++nt) {
        int n = nt * 16 + r16;
        float bb = b_kj[n];
#pragma unroll
        for (int v = 0; v < 8; ++v) {
            int r = wave * 16 + v + 8 * hh;   // block-local row
            float re = 0.0f;
#pragma unroll
            for (int j = 0; j < NR_DIM; ++j) re += sRbf[r * NR_DIM + j] * sWr[j * 128 + n];
            float t = silu_f(acc[nt][v] + bb) * re;
            sX[r * 128 + n] = (__bf16)t;      // per-wave-private slice: no race
        }
    }
    __syncthreads();

    // ---- GEMM 3: t @ W_down -> x_kj_down (N = 64) ----
    v8f acc3[4];
    zero_acc(acc3, 4);
    mm_tile<128, 4>(Aslice, 128, Wdn, 64, acc3);
#pragma unroll
    for (int nt = 0; nt < 4; ++nt) {
        int n = nt * 16 + r16;
#pragma unroll
        for (int v = 0; v < 8; ++v) {
            int ge = e0 + wave * 16 + v + 8 * hh;
            if (ge < E) xkj_down_out[(size_t)ge * I_DIM + n] = silu_f(acc3[nt][v]);
        }
    }
}

// ---------------------------------------------------------------------------
// triplet: sbf embedding (42 -> 8 -> 64, factored), gather x_kj_down[idx_kj],
// multiply, scatter-add into agg[idx_ji]. Memory/atomic bound -> VALU + L2 atomics.
// 256 threads, 256 triplets per block, sbf block staged via LDS for coalescing.
// ---------------------------------------------------------------------------
__global__ void triplet_kernel(const float* __restrict__ sbf,
                               const int* __restrict__ idx_kj,
                               const int* __restrict__ idx_ji,
                               const float* __restrict__ Ws1,   // [42,8]
                               const float* __restrict__ Ws2,   // [8,64]
                               const float* __restrict__ xkj_down,
                               float* __restrict__ agg,
                               int T) {
    __shared__ float sS[256 * NSR_DIM];
    __shared__ float sW1[NSR_DIM * B_DIM];
    __shared__ float sW2[B_DIM * I_DIM];

    const int t0 = blockIdx.x * 256;
    for (int i = threadIdx.x; i < NSR_DIM * B_DIM; i += 256) sW1[i] = Ws1[i];
    for (int i = threadIdx.x; i < B_DIM * I_DIM; i += 256) sW2[i] = Ws2[i];

    int nrow = T - t0; if (nrow > 256) nrow = 256;
    const float* src = sbf + (size_t)t0 * NSR_DIM;
    for (int i = threadIdx.x; i < nrow * NSR_DIM; i += 256) sS[i] = src[i];
    __syncthreads();

    const int t = t0 + threadIdx.x;
    if (t >= T) return;

    const int kj = idx_kj[t];
    const int ji = idx_ji[t];
    const float4* g4 = (const float4*)(xkj_down + (size_t)kj * I_DIM);
    float* dst = agg + (size_t)ji * I_DIM;
    __builtin_prefetch(g4, 0, 1);   // global_prefetch for the gather row

    // p = sbf_row @ W_sbf1   (42 x 8)
    float p[B_DIM];
#pragma unroll
    for (int b = 0; b < B_DIM; ++b) p[b] = 0.0f;
    const float* row = sS + threadIdx.x * NSR_DIM;
#pragma unroll
    for (int k = 0; k < NSR_DIM; ++k) {
        float rv = row[k];
#pragma unroll
        for (int b = 0; b < B_DIM; ++b) p[b] += rv * sW1[k * B_DIM + b];
    }

    // m = gathered * (p @ W_sbf2); scatter-add
#pragma unroll
    for (int i = 0; i < I_DIM / 4; ++i) {
        float4 xv = g4[i];
        float s0 = 0.f, s1 = 0.f, s2 = 0.f, s3 = 0.f;
#pragma unroll
        for (int b = 0; b < B_DIM; ++b) {
            const float* w = sW2 + b * I_DIM + i * 4;
            s0 += p[b] * w[0]; s1 += p[b] * w[1];
            s2 += p[b] * w[2]; s3 += p[b] * w[3];
        }
        atomicAdd(dst + i * 4 + 0, xv.x * s0);
        atomicAdd(dst + i * 4 + 1, xv.y * s1);
        atomicAdd(dst + i * 4 + 2, xv.z * s2);
        atomicAdd(dst + i * 4 + 3, xv.w * s3);
    }
}

// ---------------------------------------------------------------------------
// edge_back: h = x_ji + silu(agg@W_up); residual_before; silu(h@W_lin+b)+x;
// residual_after x2; write out. 8 WMMA GEMMs fused, h kept in f32 registers
// (C-layout), round-tripped through bf16 LDS for next GEMM's A fragments.
// 128 threads (4 waves), 64 edges per block.
// ---------------------------------------------------------------------------
__global__ void edge_back_kernel(const float* __restrict__ x,
                                 const float* __restrict__ x_ji,
                                 const float* __restrict__ agg,
                                 const __bf16* __restrict__ Wup,    // [64,128]
                                 const __bf16* __restrict__ Wrb,    // [2,128,128]
                                 const __bf16* __restrict__ Wlin,   // [128,128]
                                 const __bf16* __restrict__ Wra,    // [4,128,128]
                                 const float* __restrict__ b_rb,    // [2,128]
                                 const float* __restrict__ b_lin,   // [128]
                                 const float* __restrict__ b_ra,    // [4,128]
                                 float* __restrict__ out,
                                 int E) {
    __shared__ __bf16 sH[64 * 128];
    __shared__ __bf16 sT[64 * 128];

    const int tid  = threadIdx.x;
    const int e0   = blockIdx.x * 64;
    const int wave = tid >> 5, lane = tid & 31;
    const int r16  = lane & 15, hh = lane >> 4;

    // stage agg tile (64 x 64) into sT (lda = 64)
    for (int i = tid; i < 64 * 64; i += 128) {
        int r = i >> 6, c = i & 63;
        int ge = e0 + r;
        sT[i] = (__bf16)((ge < E) ? agg[(size_t)ge * I_DIM + c] : 0.0f);
    }
    __syncthreads();

    int grow[8];
#pragma unroll
    for (int v = 0; v < 8; ++v) grow[v] = e0 + wave * 16 + v + 8 * hh;

    float hreg[8][8];

    // ---- u = silu(agg @ W_up); h = x_ji + u ----
    {
        v8f acc[8];
        zero_acc(acc, 8);
        mm_tile<64, 8>(sT + wave * 16 * 64, 64, Wup, 128, acc);
#pragma unroll
        for (int nt = 0; nt < 8; ++nt) {
            int n = nt * 16 + r16;
#pragma unroll
            for (int v = 0; v < 8; ++v) {
                float xj = (grow[v] < E) ? x_ji[(size_t)grow[v] * H_DIM + n] : 0.0f;
                hreg[nt][v] = xj + silu_f(acc[nt][v]);
            }
        }
    }
    __syncthreads();

    // write h -> sH (bf16, per-wave-private slice)
#pragma unroll
    for (int nt = 0; nt < 8; ++nt) {
        int n = nt * 16 + r16;
#pragma unroll
        for (int v = 0; v < 8; ++v) {
            int r = wave * 16 + v + 8 * hh;
            sH[r * 128 + n] = (__bf16)hreg[nt][v];
        }
    }
    __syncthreads();

    // residual layer: h = h + silu(silu(h@W0+b0) @ W1 + b1)
    auto res_layer = [&](const __bf16* W0, const __bf16* W1,
                         const float* bb0, const float* bb1) {
        v8f a1[8];
        zero_acc(a1, 8);
        mm_tile<128, 8>(sH + wave * 16 * 128, 128, W0, 128, a1);
#pragma unroll
        for (int nt = 0; nt < 8; ++nt) {
            int n = nt * 16 + r16;
            float bb = bb0[n];
#pragma unroll
            for (int v = 0; v < 8; ++v) {
                int r = wave * 16 + v + 8 * hh;
                sT[r * 128 + n] = (__bf16)silu_f(a1[nt][v] + bb);
            }
        }
        __syncthreads();
        v8f a2[8];
        zero_acc(a2, 8);
        mm_tile<128, 8>(sT + wave * 16 * 128, 128, W1, 128, a2);
#pragma unroll
        for (int nt = 0; nt < 8; ++nt) {
            int n = nt * 16 + r16;
            float bb = bb1[n];
#pragma unroll
            for (int v = 0; v < 8; ++v) {
                hreg[nt][v] += silu_f(a2[nt][v] + bb);
                int r = wave * 16 + v + 8 * hh;
                sH[r * 128 + n] = (__bf16)hreg[nt][v];
            }
        }
        __syncthreads();
    };

    // residual-before (1 layer)
    res_layer(Wrb, Wrb + 16384, b_rb, b_rb + 128);

    // h = silu(h @ W_lin + b_lin) + x   (skip connection)
    {
        v8f a3[8];
        zero_acc(a3, 8);
        mm_tile<128, 8>(sH + wave * 16 * 128, 128, Wlin, 128, a3);
#pragma unroll
        for (int nt = 0; nt < 8; ++nt) {
            int n = nt * 16 + r16;
            float bb = b_lin[n];
#pragma unroll
            for (int v = 0; v < 8; ++v) {
                float xv = (grow[v] < E) ? x[(size_t)grow[v] * H_DIM + n] : 0.0f;
                hreg[nt][v] = silu_f(a3[nt][v] + bb) + xv;
                int r = wave * 16 + v + 8 * hh;
                sH[r * 128 + n] = (__bf16)hreg[nt][v];
            }
        }
        __syncthreads();
    }

    // residual-after (2 layers)
    res_layer(Wra,             Wra + 16384,     b_ra,           b_ra + 128);
    res_layer(Wra + 2 * 16384, Wra + 3 * 16384, b_ra + 2 * 128, b_ra + 3 * 128);

    // store final h
#pragma unroll
    for (int nt = 0; nt < 8; ++nt) {
        int n = nt * 16 + r16;
#pragma unroll
        for (int v = 0; v < 8; ++v)
            if (grow[v] < E) out[(size_t)grow[v] * H_DIM + n] = hreg[nt][v];
    }
}

// ---------------------------------------------------------------------------
// Launch
// ---------------------------------------------------------------------------
extern "C" void kernel_launch(void* const* d_in, const int* in_sizes, int n_in,
                              void* d_out, int out_size, void* d_ws, size_t ws_size,
                              hipStream_t stream) {
    const int E = E_EDGES, T = T_TRIP;

    const float* x       = (const float*)d_in[0];
    const float* rbf     = (const float*)d_in[1];
    const float* sbf     = (const float*)d_in[2];
    const int*   idx_kj  = (const int*)d_in[3];
    const int*   idx_ji  = (const int*)d_in[4];
    const float* W_rbf1  = (const float*)d_in[5];
    const float* W_rbf2  = (const float*)d_in[6];
    const float* W_sbf1  = (const float*)d_in[7];
    const float* W_sbf2  = (const float*)d_in[8];
    const float* W_kj    = (const float*)d_in[9];
    const float* b_kj    = (const float*)d_in[10];
    const float* W_ji    = (const float*)d_in[11];
    const float* b_ji    = (const float*)d_in[12];
    const float* W_down  = (const float*)d_in[13];
    const float* W_up    = (const float*)d_in[14];
    const float* W_rb    = (const float*)d_in[15];
    const float* b_rb    = (const float*)d_in[16];
    const float* W_lin   = (const float*)d_in[17];
    const float* b_lin   = (const float*)d_in[18];
    const float* W_ra    = (const float*)d_in[19];
    const float* b_ra    = (const float*)d_in[20];

    // workspace layout (all 256B-aligned offsets)
    char* ws = (char*)d_ws;
    float* x_ji_buf = (float*)ws;  ws += (size_t)E * H_DIM * 4;   // 102.4 MB
    float* xkj_down = (float*)ws;  ws += (size_t)E * I_DIM * 4;   //  51.2 MB
    float* agg      = (float*)ws;  ws += (size_t)E * I_DIM * 4;   //  51.2 MB
    float* Wrbfc    = (float*)ws;  ws += NR_DIM * H_DIM * 4;
    __bf16* Wji16 = (__bf16*)ws;   ws += 16384 * 2;
    __bf16* Wkj16 = (__bf16*)ws;   ws += 16384 * 2;
    __bf16* Wdn16 = (__bf16*)ws;   ws += 8192 * 2;
    __bf16* Wup16 = (__bf16*)ws;   ws += 8192 * 2;
    __bf16* Wrb16 = (__bf16*)ws;   ws += 32768 * 2;
    __bf16* Wlin16 = (__bf16*)ws;  ws += 16384 * 2;
    __bf16* Wra16 = (__bf16*)ws;   ws += 65536 * 2;

    // weight conversions f32 -> bf16
    auto cvt = [&](const float* s, __bf16* d, int n) {
        cvt_bf16_kernel<<<(n + 255) / 256, 256, 0, stream>>>(s, d, n);
    };
    cvt(W_ji,  Wji16,  16384);
    cvt(W_kj,  Wkj16,  16384);
    cvt(W_down, Wdn16, 8192);
    cvt(W_up,  Wup16,  8192);
    cvt(W_rb,  Wrb16,  32768);
    cvt(W_lin, Wlin16, 16384);
    cvt(W_ra,  Wra16,  65536);

    combine_rbf_kernel<<<3, 256, 0, stream>>>(W_rbf1, W_rbf2, Wrbfc);

    hipMemsetAsync(agg, 0, (size_t)E * I_DIM * 4, stream);

    const int edge_blocks = (E + 63) / 64;
    edge_front_kernel<<<edge_blocks, 128, 0, stream>>>(
        x, rbf, Wji16, Wkj16, Wdn16, b_ji, b_kj, Wrbfc,
        x_ji_buf, xkj_down, E);

    triplet_kernel<<<(T + 255) / 256, 256, 0, stream>>>(
        sbf, idx_kj, idx_ji, W_sbf1, W_sbf2, xkj_down, agg, T);

    edge_back_kernel<<<edge_blocks, 128, 0, stream>>>(
        x, x_ji_buf, agg, Wup16, Wrb16, Wlin16, Wra16,
        b_rb, b_lin, b_ra, (float*)d_out, E);
}
